// Generator_44229573214820
// MI455X (gfx1250) — compile-verified, matrix operand order checked by reading, not run on hardware
//
#include <hip/hip_runtime.h>
#include <hip/hip_bf16.h>

typedef __attribute__((ext_vector_type(16))) _Float16 v16h;
typedef __attribute__((ext_vector_type(8)))  float    v8f;
typedef __attribute__((ext_vector_type(4)))  float    v4f;

// ---------------- problem constants ----------------
#define T_IN   4096
#define C_IN   8
#define HS     32
#define NF     3
#define T1     1364          // (4096-7)/3+1
#define T2     680           // (1364-5)/2+1
#define T2P    688           // padded to 43 WMMA row tiles
#define NTILE  43            // 688/16
#define OUT_N  16777216L     // 4096*4096

// ---------------- workspace layout (floats) ----------------
#define OFF_STATS 0                  // 2 floats: sum, sumsq
#define OFF_PROJ  16                 // 32
#define OFF_VEC1  64                 // 4096
#define OFF_VEC2  (64 + 4096)        // 4096
#define OFF_EXT   8256
#define SZ_Z1     (16 * T1)          // 21824
#define SZ_BUF    (T2P * 64)         // 44032
#define SZ_G      (NTILE * 2048)     // 88064: G in WMMA D-native tiled layout
#define PER_EXT   (SZ_Z1 + 2 * SZ_BUF + 2 * SZ_G)

// ---------------- small helpers ----------------
__device__ __forceinline__ float sigf(float x) { return 1.f / (1.f + __expf(-x)); }

// Fill halves frag[e0..e0+7] from two contiguous float4s (one 8-float run).
__device__ __forceinline__ void pack8(v16h& frag, int e0, v4f p0, v4f p1) {
  frag[e0 + 0] = (_Float16)p0.x; frag[e0 + 1] = (_Float16)p0.y;
  frag[e0 + 2] = (_Float16)p0.z; frag[e0 + 3] = (_Float16)p0.w;
  frag[e0 + 4] = (_Float16)p1.x; frag[e0 + 5] = (_Float16)p1.y;
  frag[e0 + 6] = (_Float16)p1.z; frag[e0 + 7] = (_Float16)p1.w;
}

__device__ __forceinline__ float conv3x3_val(const float* __restrict__ v1,
                                             const float* __restrict__ v2,
                                             const float* __restrict__ w,
                                             int y, int x) {
  float a0 = (y > 0)    ? v1[y - 1] : 0.f;
  float a1 = v1[y];
  float a2 = (y < 4095) ? v1[y + 1] : 0.f;
  float b0 = (x > 0)    ? v2[x - 1] : 0.f;
  float b1 = v2[x];
  float b2 = (x < 4095) ? v2[x + 1] : 0.f;
  return a0 * (w[0] * b0 + w[1] * b1 + w[2] * b2) +
         a1 * (w[3] * b0 + w[4] * b1 + w[5] * b2) +
         a2 * (w[6] * b0 + w[7] * b1 + w[8] * b2);
}

// ============================================================
// 0) zero the BN accumulators (fresh every call)
// ============================================================
__global__ void k_init(float* ws) {
  if (threadIdx.x < 2) ws[OFF_STATS + threadIdx.x] = 0.f;
}

// ============================================================
// 1) conv1+BN1+ReLU, conv2+BN2+ReLU  (one block per extractor)
//    writes bufA = X0 (688 x 64), cols 0..15 = features, rest 0
// ============================================================
__global__ __launch_bounds__(256)
void k_front(const float* __restrict__ x0, const float* __restrict__ x1,
             const float* __restrict__ x2,
             const float* __restrict__ w1, const float* __restrict__ g1,
             const float* __restrict__ b1,
             const float* __restrict__ w2, const float* __restrict__ g2,
             const float* __restrict__ b2, float* ws) {
  const int i = blockIdx.x;
  const float* x = (i == 0) ? x0 : ((i == 1) ? x1 : x2);
  float* E  = ws + OFF_EXT + (long)i * PER_EXT;
  float* z1 = E;                 // [16][T1]
  float* X0 = E + SZ_Z1;         // bufA [T2P][64]
  const float* W1 = w1 + i * 16 * 8 * 7;
  const float* W2 = w2 + i * 16 * 16 * 5;

  __shared__ float ps[256], ps2[256], chm[16], chi[16];

  // conv1 (stride 3, k=7)
  for (int idx = threadIdx.x; idx < 16 * T1; idx += 256) {
    int c = idx / T1, t = idx % T1;
    const float* wr = W1 + c * 56;   // [ci][k]
    float acc = 0.f;
    int base = 3 * t;
    for (int ci = 0; ci < 8; ++ci)
      for (int k = 0; k < 7; ++k)
        acc += wr[ci * 7 + k] * x[(base + k) * 8 + ci];
    z1[c * T1 + t] = acc;
  }
  __syncthreads();
  // BN1 stats (train mode over time, biased var, eps=1e-3)
  {
    int c = threadIdx.x >> 4, l = threadIdx.x & 15;
    float s = 0.f, s2 = 0.f;
    for (int t = l; t < T1; t += 16) { float v = z1[c * T1 + t]; s += v; s2 += v * v; }
    ps[threadIdx.x] = s; ps2[threadIdx.x] = s2;
  }
  __syncthreads();
  if (threadIdx.x < 16) {
    int c = threadIdx.x;
    float s = 0.f, s2 = 0.f;
    for (int l = 0; l < 16; ++l) { s += ps[c * 16 + l]; s2 += ps2[c * 16 + l]; }
    float m = s / (float)T1, v = s2 / (float)T1 - m * m;
    chm[c] = m; chi[c] = rsqrtf(v + 1e-3f);
  }
  __syncthreads();
  for (int idx = threadIdx.x; idx < 16 * T1; idx += 256) {
    int c = idx / T1;
    float v = (z1[idx] - chm[c]) * chi[c] * g1[i * 16 + c] + b1[i * 16 + c];
    z1[idx] = v > 0.f ? v : 0.f;
  }
  __syncthreads();
  // conv2 (stride 2, k=5) -> raw into X0[t][c]
  for (int idx = threadIdx.x; idx < 16 * T2; idx += 256) {
    int c = idx / T2, t = idx % T2;
    const float* wr = W2 + c * 80;   // [ci][k]
    float acc = 0.f;
    int base = 2 * t;
    for (int ci = 0; ci < 16; ++ci)
      for (int k = 0; k < 5; ++k)
        acc += wr[ci * 5 + k] * z1[ci * T1 + base + k];
    X0[t * 64 + c] = acc;
  }
  __syncthreads();
  // BN2 stats
  {
    int c = threadIdx.x >> 4, l = threadIdx.x & 15;
    float s = 0.f, s2 = 0.f;
    for (int t = l; t < T2; t += 16) { float v = X0[t * 64 + c]; s += v; s2 += v * v; }
    ps[threadIdx.x] = s; ps2[threadIdx.x] = s2;
  }
  __syncthreads();
  if (threadIdx.x < 16) {
    int c = threadIdx.x;
    float s = 0.f, s2 = 0.f;
    for (int l = 0; l < 16; ++l) { s += ps[c * 16 + l]; s2 += ps2[c * 16 + l]; }
    float m = s / (float)T2, v = s2 / (float)T2 - m * m;
    chm[c] = m; chi[c] = rsqrtf(v + 1e-3f);
  }
  __syncthreads();
  // normalize+relu, zero cols 16..63, zero pad rows
  for (int idx = threadIdx.x; idx < T2 * 64; idx += 256) {
    int c = idx & 63;
    float v = 0.f;
    if (c < 16) {
      v = (X0[idx] - chm[c]) * chi[c] * g2[i * 16 + c] + b2[i * 16 + c];
      v = v > 0.f ? v : 0.f;
    }
    X0[idx] = v;
  }
  for (int idx = T2 * 64 + threadIdx.x; idx < T2P * 64; idx += 256) X0[idx] = 0.f;
}

// ============================================================
// 2) one biLSTM layer: WMMA gate GEMM + serial recurrence
//    grid: 6 blocks = (extractor, direction), 256 threads
//    G is kept in the WMMA D-native layout
//      G[tile][wave][lane][r]  (lane = half*16 + n, r = 0..7)
//    so the GEMM stores are two unconditional b128 per tile.
// ============================================================
template <int LAYER>
__global__ __launch_bounds__(256)
void k_lstm(const float* __restrict__ l0_Wih, const float* __restrict__ l0_Whh,
            const float* __restrict__ l0_b,
            const float* __restrict__ l12_Wih, const float* __restrict__ l12_Whh,
            const float* __restrict__ l12_b, float* ws) {
  constexpr int IN  = (LAYER == 0) ? 16 : 64;
  constexpr int KFR = (LAYER == 0) ? 1 : 2;

  const int i = blockIdx.x >> 1, dir = blockIdx.x & 1;
  float* E    = ws + OFF_EXT + (long)i * PER_EXT;
  float* bufA = E + SZ_Z1;
  float* bufB = bufA + SZ_BUF;
  float* G    = bufB + SZ_BUF + dir * SZ_G;
  const float* X = (LAYER == 1) ? bufB : bufA;
  float* Xout    = (LAYER == 1) ? bufA : bufB;

  const float *Wih, *Whh, *bias;
  if (LAYER == 0) {
    int o = i * 2 + dir;
    Wih = l0_Wih + o * 128 * 16; Whh = l0_Whh + o * 128 * 32; bias = l0_b + o * 128;
  } else {
    int o = (i * 2 + (LAYER - 1)) * 2 + dir;
    Wih = l12_Wih + o * 128 * 64; Whh = l12_Whh + o * 128 * 32; bias = l12_b + o * 128;
  }

  __shared__ float WhhT[32][128];   // transposed for conflict-free recurrence reads
  for (int idx = threadIdx.x; idx < 128 * 32; idx += 256) {
    int g = idx >> 5, j = idx & 31;
    WhhT[j][g] = Whh[g * 32 + j];
  }

  const int lane = threadIdx.x & 31, wave = threadIdx.x >> 5;
  const int half = lane >> 4, nn = lane & 15;
  const int nbase = wave * 16;           // wave owns one 16-gate N-tile

  // ---- B fragments (Wih^T), loop-invariant: build once per wave ----
  // ISA layout: element e<8 -> k = 32*kk + 8*half + e ; e>=8 -> k = +16.
  v16h bf[KFR];
  {
    const float* Wrow = Wih + (nbase + nn) * IN;
    for (int kk = 0; kk < KFR; ++kk) {
      const v4f* wr4 = (const v4f*)(Wrow + kk * 32 + half * 8);
      pack8(bf[kk], 0, wr4[0], wr4[1]);
      if (IN == 16) {
        for (int e = 8; e < 16; ++e) bf[kk][e] = (_Float16)0.f;  // k>=16 is zero
      } else {
        pack8(bf[kk], 8, wr4[4], wr4[5]);                        // +16 floats
      }
    }
  }

  // ---- gate GEMM: G(tile-layout) = X @ Wih^T  (bias added in recurrence) ----
  for (int tt = 0; tt < NTILE; ++tt) {
    const int tbase = tt << 4;
    const int m = tbase + nn;            // A: lanes 0-15 / 16-31 both map M=lane%16
    const float* Xr = X + m * 64;        // X zero-padded to 688x64
    v8f acc = {};
    for (int kk = 0; kk < KFR; ++kk) {
      const v4f* xr4 = (const v4f*)(Xr + kk * 32 + half * 8);
      v16h a;
      pack8(a, 0, xr4[0], xr4[1]);
      pack8(a, 8, xr4[4], xr4[5]);
      acc = __builtin_amdgcn_wmma_f32_16x16x32_f16(false, a, false, bf[kk],
                                                   (short)0, acc, false, false);
    }
    // D-native store: contiguous 8 floats per lane, no guards
    float* gp = G + (((tt * 8) + wave) * 32 + lane) * 8;
    v4f lo = {acc[0], acc[1], acc[2], acc[3]};
    v4f hi = {acc[4], acc[5], acc[6], acc[7]};
    *(v4f*)gp       = lo;
    *(v4f*)(gp + 4) = hi;
  }
  __syncthreads();

  // ---- serial recurrence on wave 0; others zero the pad rows ----
  if (wave == 0) {
    // gate g = lane + 32q at time t lives at:
    //   (t>>4)*2048 + (2q + (lane>>4))*256 + ((t>>3)&1)*128 + (lane&15)*8 + (t&7)
    const int gbase = ((lane >> 4) << 8) + ((lane & 15) << 3);
    const float bl0 = bias[lane],      bl1 = bias[32 + lane];
    const float bl2 = bias[64 + lane], bl3 = bias[96 + lane];
    float c = 0.f, h = 0.f;   // lane owns hidden index `lane`
    for (int s = 0; s < T2; ++s) {
      int t = dir ? (T2 - 1 - s) : s;
      const float* Gt = G + ((t >> 4) << 11) + (((t >> 3) & 1) << 7) + (t & 7) + gbase;
      int tn = dir ? (t > 0 ? t - 1 : 0) : (t < T2 - 1 ? t + 1 : t);
      __builtin_prefetch((const void*)(G + ((tn >> 4) << 11)), 0, 1);
      float g0 = Gt[0]    + bl0;
      float g1 = Gt[512]  + bl1;
      float g2 = Gt[1024] + bl2;
      float g3 = Gt[1536] + bl3;
      for (int j = 0; j < 32; ++j) {
        float hj = __shfl(h, j, 32);
        g0 += hj * WhhT[j][lane];
        g1 += hj * WhhT[j][32 + lane];
        g2 += hj * WhhT[j][64 + lane];
        g3 += hj * WhhT[j][96 + lane];
      }
      float ig = sigf(g0), fg = sigf(g1), gg = tanhf(g2), og = sigf(g3);
      c = fg * c + ig * gg;
      h = og * tanhf(c);
      Xout[t * 64 + dir * 32 + lane] = h;
    }
  } else {
    for (int idx = threadIdx.x - 32; idx < 8 * 64; idx += 224)
      Xout[T2 * 64 + idx] = 0.f;
  }
}

// ============================================================
// 3) fc projection + 4096-step dual LSTMCell scan -> vec1/vec2
//    1 block, 64 threads: wave 0 = cell1, wave 1 = cell2
// ============================================================
__global__ __launch_bounds__(64)
void k_scan(const float* __restrict__ fc_W, const float* __restrict__ fc_b,
            const float* __restrict__ c1_Wih, const float* __restrict__ c1_Whh,
            const float* __restrict__ c1_b,
            const float* __restrict__ c2_Wih, const float* __restrict__ c2_Whh,
            const float* __restrict__ c2_b,
            const float* __restrict__ tr_W, const float* __restrict__ tr_b,
            const float* __restrict__ h1_0, const float* __restrict__ c1_0,
            const float* __restrict__ h2_0, const float* __restrict__ c2_0,
            float* ws) {
  __shared__ float WcT[2][32][128];   // (Wih + Whh)^T — valid for steps t>=1 where x==h_prev
  __shared__ float proj[32];
  const int tid = threadIdx.x, wave = tid >> 5, lane = tid & 31;

  for (int idx = tid; idx < 2 * 128 * 32; idx += 64) {
    int cell = idx >> 12, r = idx & 4095, g = r >> 5, j = r & 31;
    const float* Wi = cell ? c2_Wih : c1_Wih;
    const float* Wh = cell ? c2_Whh : c1_Whh;
    WcT[cell][j][g] = Wi[g * 32 + j] + Wh[g * 32 + j];
  }
  if (tid < 32) {   // proj = feats @ fc_W.T + fc_b ; feats = last biLSTM row per extractor
    float acc = fc_b[tid];
    for (int j = 0; j < 192; ++j) {
      int i = j / 64, cc = j % 64;
      float f = ws[OFF_EXT + (long)i * PER_EXT + SZ_Z1 + SZ_BUF + 679 * 64 + cc]; // bufB
      acc += fc_W[tid * 192 + j] * f;
    }
    proj[tid] = acc;
    ws[OFF_PROJ + tid] = acc;
  }
  __syncthreads();

  const float* Wih = wave ? c2_Wih : c1_Wih;
  const float* Whh = wave ? c2_Whh : c1_Whh;
  const float* bb  = wave ? c2_b   : c1_b;
  float b0 = bb[lane], b1 = bb[32 + lane], b2 = bb[64 + lane], b3 = bb[96 + lane];
  float h = (wave ? h2_0 : h1_0)[lane];
  float c = (wave ? c2_0 : c1_0)[lane];
  float* vec = ws + (wave ? OFF_VEC2 : OFF_VEC1);
  const float trw = tr_W[lane], trb = tr_b[0];
  float x = proj[lane];

  // step 0: x = proj, h = h*_0 (distinct)
  {
    float g0 = b0, g1 = b1, g2 = b2, g3 = b3;
    for (int j = 0; j < 32; ++j) {
      float xj = __shfl(x, j, 32);
      float hj = __shfl(h, j, 32);
      g0 += xj * Wih[lane * 32 + j]        + hj * Whh[lane * 32 + j];
      g1 += xj * Wih[(32 + lane) * 32 + j] + hj * Whh[(32 + lane) * 32 + j];
      g2 += xj * Wih[(64 + lane) * 32 + j] + hj * Whh[(64 + lane) * 32 + j];
      g3 += xj * Wih[(96 + lane) * 32 + j] + hj * Whh[(96 + lane) * 32 + j];
    }
    c = sigf(g1) * c + sigf(g0) * tanhf(g2);
    h = sigf(g3) * tanhf(c);
    float r = h * trw;
    for (int off = 16; off; off >>= 1) r += __shfl_xor(r, off, 32);
    if (lane == 0) vec[0] = r + trb;
  }
  // steps 1..4095: x == h_prev  =>  gates = b + h · (Wih+Whh)^T
  for (int t = 1; t < T_IN; ++t) {
    float g0 = b0, g1 = b1, g2 = b2, g3 = b3;
    for (int j = 0; j < 32; ++j) {
      float hj = __shfl(h, j, 32);
      g0 += hj * WcT[wave][j][lane];
      g1 += hj * WcT[wave][j][32 + lane];
      g2 += hj * WcT[wave][j][64 + lane];
      g3 += hj * WcT[wave][j][96 + lane];
    }
    c = sigf(g1) * c + sigf(g0) * tanhf(g2);
    h = sigf(g3) * tanhf(c);
    float r = h * trw;
    for (int off = 16; off; off >>= 1) r += __shfl_xor(r, off, 32);
    if (lane == 0) vec[t] = r + trb;
  }
}

// ============================================================
// 4) BN statistics over the (never-materialized) 4096x4096 map
// ============================================================
__global__ __launch_bounds__(256)
void k_stats(const float* __restrict__ ws_ro, const float* __restrict__ ref_w,
             float* ws) {
  const float* v1 = ws_ro + OFF_VEC1;
  const float* v2 = ws_ro + OFF_VEC2;
  float w[9];
  for (int q = 0; q < 9; ++q) w[q] = ref_w[q];
  float s = 0.f, s2 = 0.f;
  for (long idx = (long)blockIdx.x * 256 + threadIdx.x; idx < OUT_N;
       idx += (long)gridDim.x * 256) {
    int y = (int)(idx >> 12), xx = (int)(idx & 4095);
    float val = conv3x3_val(v1, v2, w, y, xx);
    s += val; s2 += val * val;
  }
  __shared__ float rs[256], rs2[256];
  rs[threadIdx.x] = s; rs2[threadIdx.x] = s2;
  __syncthreads();
  for (int off = 128; off; off >>= 1) {
    if (threadIdx.x < off) {
      rs[threadIdx.x]  += rs[threadIdx.x + off];
      rs2[threadIdx.x] += rs2[threadIdx.x + off];
    }
    __syncthreads();
  }
  if (threadIdx.x == 0) {
    atomicAdd(&ws[OFF_STATS + 0], rs[0]);
    atomicAdd(&ws[OFF_STATS + 1], rs2[0]);
  }
}

// ============================================================
// 5) recompute conv value, apply BN+ReLU, write output
// ============================================================
__global__ __launch_bounds__(256)
void k_final(const float* __restrict__ ws, const float* __restrict__ ref_w,
             const float* __restrict__ ref_g, const float* __restrict__ ref_b,
             float* __restrict__ out) {
  const float* v1 = ws + OFF_VEC1;
  const float* v2 = ws + OFF_VEC2;
  float w[9];
  for (int q = 0; q < 9; ++q) w[q] = ref_w[q];
  const float inv_n = 1.f / (float)OUT_N;
  const float mean = ws[OFF_STATS + 0] * inv_n;
  const float var  = ws[OFF_STATS + 1] * inv_n - mean * mean;
  const float scale = ref_g[0] * rsqrtf(var + 1e-3f);
  const float shift = ref_b[0];
  for (long idx = (long)blockIdx.x * 256 + threadIdx.x; idx < OUT_N;
       idx += (long)gridDim.x * 256) {
    int y = (int)(idx >> 12), xx = (int)(idx & 4095);
    float val = conv3x3_val(v1, v2, w, y, xx);
    float r = (val - mean) * scale + shift;
    out[idx] = r > 0.f ? r : 0.f;
  }
}

// ============================================================
extern "C" void kernel_launch(void* const* d_in, const int* in_sizes, int n_in,
                              void* d_out, int out_size, void* d_ws, size_t ws_size,
                              hipStream_t stream) {
  const float* x0      = (const float*)d_in[0];
  const float* x1      = (const float*)d_in[1];
  const float* x2      = (const float*)d_in[2];
  const float* conv1_w = (const float*)d_in[3];
  const float* bn1_g   = (const float*)d_in[4];
  const float* bn1_b   = (const float*)d_in[5];
  const float* conv2_w = (const float*)d_in[6];
  const float* bn2_g   = (const float*)d_in[7];
  const float* bn2_b   = (const float*)d_in[8];
  const float* l0_Wih  = (const float*)d_in[9];
  const float* l0_Whh  = (const float*)d_in[10];
  const float* l0_b    = (const float*)d_in[11];
  const float* l12_Wih = (const float*)d_in[12];
  const float* l12_Whh = (const float*)d_in[13];
  const float* l12_b   = (const float*)d_in[14];
  const float* fc_W    = (const float*)d_in[15];
  const float* fc_b    = (const float*)d_in[16];
  const float* c1_Wih  = (const float*)d_in[17];
  const float* c1_Whh  = (const float*)d_in[18];
  const float* c1_b    = (const float*)d_in[19];
  const float* c2_Wih  = (const float*)d_in[20];
  const float* c2_Whh  = (const float*)d_in[21];
  const float* c2_b    = (const float*)d_in[22];
  const float* tr_W    = (const float*)d_in[23];
  const float* tr_b    = (const float*)d_in[24];
  const float* ref_w   = (const float*)d_in[25];
  const float* ref_g   = (const float*)d_in[26];
  const float* ref_b   = (const float*)d_in[27];
  const float* h1_0    = (const float*)d_in[28];
  const float* c1_0    = (const float*)d_in[29];
  const float* h2_0    = (const float*)d_in[30];
  const float* c2_0    = (const float*)d_in[31];
  float* ws  = (float*)d_ws;
  float* out = (float*)d_out;

  k_init<<<1, 64, 0, stream>>>(ws);
  k_front<<<NF, 256, 0, stream>>>(x0, x1, x2, conv1_w, bn1_g, bn1_b,
                                  conv2_w, bn2_g, bn2_b, ws);
  k_lstm<0><<<NF * 2, 256, 0, stream>>>(l0_Wih, l0_Whh, l0_b,
                                        l12_Wih, l12_Whh, l12_b, ws);
  k_lstm<1><<<NF * 2, 256, 0, stream>>>(l0_Wih, l0_Whh, l0_b,
                                        l12_Wih, l12_Whh, l12_b, ws);
  k_lstm<2><<<NF * 2, 256, 0, stream>>>(l0_Wih, l0_Whh, l0_b,
                                        l12_Wih, l12_Whh, l12_b, ws);
  k_scan<<<1, 64, 0, stream>>>(fc_W, fc_b, c1_Wih, c1_Whh, c1_b,
                               c2_Wih, c2_Whh, c2_b, tr_W, tr_b,
                               h1_0, c1_0, h2_0, c2_0, ws);
  k_stats<<<1024, 256, 0, stream>>>(ws, ref_w, ws);
  k_final<<<4096, 256, 0, stream>>>(ws, ref_w, ref_g, ref_b, out);
}
